// DioraBase_43361989821049
// MI455X (gfx1250) — compile-verified, matrix operand order checked by reading, not run on hardware
//
#include <hip/hip_runtime.h>
#include <hip/hip_bf16.h>
#include <math.h>

#define TT 48
#define BB 32
#define DD 512
#define NCELLS 1176   /* TT*(TT+1)/2 */
#define EPSF 1e-8f

typedef __attribute__((ext_vector_type(16))) __bf16 v16bf;
typedef __attribute__((ext_vector_type(8)))  float  v8f;

__device__ __forceinline__ int off_of(int l) { return l * TT - (l * (l - 1)) / 2; }

__device__ __forceinline__ unsigned short f2bf(float f) {
    union { float f; unsigned int u; } c; c.f = f;
    unsigned int u = c.u + 0x7FFFu + ((c.u >> 16) & 1u);
    return (unsigned short)(u >> 16);
}
__device__ __forceinline__ float bf2f(unsigned short b) {
    union { unsigned int u; float f; } c; c.u = ((unsigned int)b) << 16;
    return c.f;
}
// k index inside a 16x32 (A) / 32x16 (B) bf16 WMMA fragment for element e (0..15)
// of a lane in half-group `half` (lane>>4), per CDNA5 ISA 7.12.2 layout tables.
__device__ __forceinline__ int kmap(int e, int half) {
    int j = e >> 1;
    int base = (j < 4) ? (2 * j) : (16 + 2 * (j - 4));
    return base + 8 * half + (e & 1);
}

union FragB {
    v16bf v;
    unsigned short s[16];
    unsigned int u[8];
    uint4 q[2];
};

__device__ __forceinline__ v8f wmma_bf16(const FragB& a, const FragB& b, v8f c) {
    return __builtin_amdgcn_wmma_f32_16x16x32_bf16(false, a.v, false, b.v, (short)0, c, false, false);
}

// ---------------------------------------------------------------------------
// Pack a (K x N) f32 weight into WMMA-fragment order bf16 tiles:
// tile(kt,nt): 32 lanes x 16 elems contiguous -> lane loads 2x b128.
// ---------------------------------------------------------------------------
__global__ void pack_weight_kernel(const float* __restrict__ W,
                                   unsigned short* __restrict__ Wp, int K, int N) {
    int ntiles = (K / 32) * (N / 16);
    int tid = blockIdx.x * blockDim.x + threadIdx.x;
    if (tid >= ntiles * 32) return;
    int tile = tid >> 5, lane = tid & 31;
    int nt16 = N / 16;
    int kt = tile / nt16, nt = tile % nt16;
    int pos = lane & 15, half = lane >> 4;
    unsigned short* dst = Wp + (size_t)tile * 512 + lane * 16;
#pragma unroll
    for (int e = 0; e < 16; ++e) {
        int k = kt * 32 + kmap(e, half);
        int n = nt * 16 + pos;
        dst[e] = f2bf(W[(size_t)k * N + n]);
    }
}

__global__ void transpose_bf16_kernel(const float* __restrict__ S,
                                      unsigned short* __restrict__ ST) {
    int tid = blockIdx.x * blockDim.x + threadIdx.x;
    if (tid >= DD * DD) return;
    int n = tid / DD, k = tid % DD;
    ST[tid] = f2bf(S[(size_t)k * DD + n]);  // ST[n][k] = S[k][n]
}

__global__ void zero2_kernel(float* __restrict__ a, float* __restrict__ b, int n) {
    int i = blockIdx.x * blockDim.x + threadIdx.x;
    if (i < n) { a[i] = 0.f; b[i] = 0.f; }
}

// Root: oh[root] = unit_norm(root_h), broadcast over batch. 512 threads.
__global__ void root_kernel(const float* __restrict__ root_h,
                            float* __restrict__ outbuf,
                            unsigned short* __restrict__ oh_bf) {
    __shared__ float red[16];
    __shared__ float s_inv;
    int tid = threadIdx.x;
    float v = root_h[tid];
    float ss = v * v;
    for (int m = 16; m >= 1; m >>= 1) ss += __shfl_xor(ss, m, 32);
    if ((tid & 31) == 0) red[tid >> 5] = ss;
    __syncthreads();
    if (tid == 0) {
        float t = 0.f;
        for (int w = 0; w < 16; ++w) t += red[w];
        s_inv = 1.f / fmaxf(sqrtf(t), EPSF);
    }
    __syncthreads();
    float val = v * s_inv;
    unsigned short bv = f2bf(val);
    for (int b = 0; b < BB; ++b) {
        size_t orow = (size_t)b * NCELLS + (NCELLS - 1);
        outbuf[orow * 1024 + 512 + tid] = val;
        oh_bf[orow * 512 + tid] = bv;
    }
}

// ---------------------------------------------------------------------------
// Leaf: h0 = unit_norm(tanh(x @ W_leaf + b_leaf)); also g_in = h0 @ S_in.
// 16 rows per WG, 256 threads (8 waves x 4 n-tiles), K=512.
// ---------------------------------------------------------------------------
__global__ void __launch_bounds__(256) leaf_kernel(
    const float* __restrict__ x, const unsigned short* __restrict__ Wlp,
    const float* __restrict__ b_leaf, const unsigned short* __restrict__ SinT,
    float* __restrict__ outbuf, unsigned short* __restrict__ ih_bf,
    unsigned short* __restrict__ g_in) {
    extern __shared__ char smem[];
    unsigned short* A = (unsigned short*)smem;                 // 16x512 bf16
    float* H = (float*)(smem + 16 * 512 * 2);                  // 16x512 f32
    float* nrm = (float*)(smem + 16 * 512 * 2 + 16 * 512 * 4); // 16 f32
    int tid = threadIdx.x;
    int r0 = blockIdx.x * 16;

    for (int idx = tid; idx < 16 * 512; idx += 256) {
        int m = idx >> 9, k = idx & 511;
        A[idx] = f2bf(x[(size_t)(r0 + m) * DD + k]);
    }
    __syncthreads();

    int lane = tid & 31, wave = tid >> 5;
    int pos = lane & 15, half = lane >> 4;
    v8f vz = {};
    v8f acc[4]; for (int q = 0; q < 4; ++q) acc[q] = vz;
    const unsigned int* A32 = (const unsigned int*)A;          // 256 dwords/row
    const uint4* bp = (const uint4*)Wlp;

    for (int kk = 0; kk < 16; ++kk) {
        FragB af;
#pragma unroll
        for (int j = 0; j < 8; ++j) {
            int dw = ((j < 4) ? j : j + 4) + 4 * half;
            af.u[j] = A32[pos * 256 + kk * 16 + dw];
        }
#pragma unroll
        for (int q = 0; q < 4; ++q) {
            int nt = wave * 4 + q;
            FragB bfr;
            int tile = kk * 32 + nt;
            bfr.q[0] = bp[(size_t)tile * 64 + lane * 2];
            bfr.q[1] = bp[(size_t)tile * 64 + lane * 2 + 1];
            acc[q] = wmma_bf16(af, bfr, acc[q]);
        }
    }
    for (int q = 0; q < 4; ++q) {
        int nt = wave * 4 + q;
        int n = nt * 16 + pos;
        float bias = b_leaf[n];
#pragma unroll
        for (int r = 0; r < 8; ++r) {
            int m = r + 8 * half;
            H[m * 512 + n] = tanhf(acc[q][r] + bias);
        }
    }
    __syncthreads();
    if (tid < 16) {
        float ss = 0.f;
        for (int j = 0; j < 512; ++j) { float v = H[tid * 512 + j]; ss += v * v; }
        nrm[tid] = 1.f / fmaxf(sqrtf(ss), EPSF);
    }
    __syncthreads();
    for (int idx = tid; idx < 16 * 512; idx += 256) {
        int m = idx >> 9, j = idx & 511;
        int grow = r0 + m, b = grow / TT, t = grow % TT;
        size_t orow = (size_t)b * NCELLS + t;
        float v = H[idx] * nrm[m];
        outbuf[orow * 1024 + j] = v;
        ih_bf[orow * 512 + j] = f2bf(v);
    }
    __syncthreads();
    // fused GEMV: g[m][j] = (sum_k H[m][k]*SinT[j][k]) * nrm[m]
    for (int idx = tid; idx < 16 * 512; idx += 256) {
        int m = idx >> 9, j = idx & 511;
        const unsigned short* st = SinT + (size_t)j * 512;
        float s = 0.f;
        for (int k = 0; k < 512; ++k) s += H[m * 512 + k] * bf2f(st[k]);
        s *= nrm[m];
        int grow = r0 + m, b = grow / TT, t = grow % TT;
        size_t orow = (size_t)b * NCELLS + t;
        g_in[orow * 512 + j] = f2bf(s);
    }
}

// ---------------------------------------------------------------------------
// g_out = ih @ S_out for all B*NCELLS rows. 16 rows/WG, K=N=512.
// ---------------------------------------------------------------------------
__global__ void __launch_bounds__(256) gout_kernel(
    const unsigned short* __restrict__ ih_bf, const unsigned short* __restrict__ Sp,
    unsigned short* __restrict__ g_out) {
    extern __shared__ char smem[];
    unsigned int* A32 = (unsigned int*)smem;   // 16 rows x 256 dwords
    int tid = threadIdx.x;
    size_t r0 = (size_t)blockIdx.x * 16;
    const unsigned int* src = (const unsigned int*)ih_bf;
    for (int idx = tid; idx < 16 * 256; idx += 256) A32[idx] = src[r0 * 256 + idx];
    __syncthreads();
    int lane = tid & 31, wave = tid >> 5, pos = lane & 15, half = lane >> 4;
    v8f vz = {};
    v8f acc[4]; for (int q = 0; q < 4; ++q) acc[q] = vz;
    const uint4* bp = (const uint4*)Sp;
    for (int kk = 0; kk < 16; ++kk) {
        FragB af;
#pragma unroll
        for (int j = 0; j < 8; ++j) {
            int dw = ((j < 4) ? j : j + 4) + 4 * half;
            af.u[j] = A32[pos * 256 + kk * 16 + dw];
        }
#pragma unroll
        for (int q = 0; q < 4; ++q) {
            int nt = wave * 4 + q;
            FragB bfr;
            int tile = kk * 32 + nt;
            bfr.q[0] = bp[(size_t)tile * 64 + lane * 2];
            bfr.q[1] = bp[(size_t)tile * 64 + lane * 2 + 1];
            acc[q] = wmma_bf16(af, bfr, acc[q]);
        }
    }
    for (int q = 0; q < 4; ++q) {
        int nt = wave * 4 + q, n = nt * 16 + pos;
#pragma unroll
        for (int r = 0; r < 8; ++r) {
            size_t row = r0 + r + 8 * half;
            g_out[row * 512 + n] = f2bf(acc[q][r]);
        }
    }
}

// ---------------------------------------------------------------------------
// Shared GEMM + weighted-reduce body for the level kernels.
// A is Npad x 1024 bf16 in LDS (MT = Npad/16 m-tiles, compile-time).
// Per k-step: load MT A-frags + 4 B-frags once, issue MT*4 WMMAs
// (A reused across n-tiles, B reused across m-tiles), then fold the
// softmax-weighted candidate reduction into 4 ds_add_f32 per lane.
// ---------------------------------------------------------------------------
template <int MT>
__device__ __forceinline__ void level_gemm_reduce(
    const unsigned int* A32, const uint4* bp, const float* bias_v,
    const float* sPr, float* sOut, int lane, int wave) {
    int pos = lane & 15, half = lane >> 4;
    v8f vz = {};
    v8f acc[MT * 4];
#pragma unroll
    for (int q = 0; q < MT * 4; ++q) acc[q] = vz;
    for (int kk = 0; kk < 32; ++kk) {
        FragB af[MT];
#pragma unroll
        for (int mt = 0; mt < MT; ++mt) {
            int arow = mt * 16 + pos;
#pragma unroll
            for (int j = 0; j < 8; ++j) {
                int dw = ((j < 4) ? j : j + 4) + 4 * half;
                af[mt].u[j] = A32[arow * 512 + kk * 16 + dw];
            }
        }
        if (kk + 1 < 32) {  // prefetch next k-step's packed-weight tiles
            int ntile = (kk + 1) * 32 + wave * 4;
            __builtin_prefetch((const void*)(bp + (size_t)ntile * 64 + lane * 2), 0, 3);
        }
#pragma unroll
        for (int j = 0; j < 4; ++j) {
            int nt = wave * 4 + j;
            FragB bfr;
            int tile = kk * 32 + nt;
            bfr.q[0] = bp[(size_t)tile * 64 + lane * 2];
            bfr.q[1] = bp[(size_t)tile * 64 + lane * 2 + 1];
#pragma unroll
            for (int mt = 0; mt < MT; ++mt)
                acc[mt * 4 + j] = wmma_bf16(af[mt], bfr, acc[mt * 4 + j]);
        }
    }
    // epilogue: relu(+bias) * softmax weight, fold m-tiles, LDS-atomic reduce
#pragma unroll
    for (int j = 0; j < 4; ++j) {
        int n = (wave * 4 + j) * 16 + pos;
        float bias = bias_v[n];
        float part = 0.f;
#pragma unroll
        for (int mt = 0; mt < MT; ++mt) {
#pragma unroll
            for (int r = 0; r < 8; ++r) {
                int m = mt * 16 + r + 8 * half;
                float v = fmaxf(acc[mt * 4 + j][r] + bias, 0.f);
                part += v * sPr[m];
            }
        }
        atomicAdd(&sOut[n], part);
    }
}

// ---------------------------------------------------------------------------
// Inside level: one WG per (b, i). Candidates n=0..N-1 (N=level).
// ---------------------------------------------------------------------------
template <int MT>
__global__ void __launch_bounds__(256) inside_level_kernel(
    int level, unsigned short* __restrict__ ih_bf, unsigned short* __restrict__ g_in,
    float* __restrict__ isc, const unsigned short* __restrict__ Wp,
    const float* __restrict__ b_in, const unsigned short* __restrict__ SinT,
    float* __restrict__ outbuf) {
    const int L = TT - level, N = level;
    constexpr int Npad = MT * 16;
    int b = blockIdx.x / L, i = blockIdx.x % L;

    extern __shared__ char smem[];
    unsigned int* A32 = (unsigned int*)smem;                  // Npad x 512 dwords
    float* sOut = (float*)(smem + (size_t)Npad * 2048);       // 512 f32
    float* sS = sOut + 512;                                   // 48
    float* sPr = sS + 48;                                     // 48
    int* sL = (int*)(sPr + 48);                               // 48
    int* sR = sL + 48;                                        // 48
    float* sMisc = (float*)(sR + 48);                         // 16
    __shared__ float red[8];

    int tid = threadIdx.x, lane = tid & 31, wave = tid >> 5;
    size_t brow = (size_t)b * NCELLS;

    if (tid < Npad) {
        if (tid < N) {
            sL[tid] = off_of(tid) + i;
            sR[tid] = off_of(level - tid - 1) + i + tid + 1;
        }
        sPr[tid] = 0.f;
    }
    for (int idx = tid; idx < 512; idx += 256) sOut[idx] = 0.f;
    __syncthreads();

    const unsigned int* ih32 = (const unsigned int*)ih_bf;
    for (int idx = tid; idx < Npad * 512; idx += 256) {
        int n = idx >> 9, o = idx & 511;
        unsigned int v = 0u;
        if (n < N) {
            int cell = (o < 256) ? sL[n] : sR[n];
            v = ih32[(brow + cell) * 256 + (o & 255)];
        }
        A32[idx] = v;
    }
    __syncthreads();

    // scores: one wave per candidate
    const unsigned int* g32 = (const unsigned int*)g_in;
    for (int n = wave; n < N; n += 8) {
        const unsigned int* grow = g32 + (brow + sL[n]) * 256;
        const unsigned int* rrow = A32 + n * 512 + 256;
        float s = 0.f;
        for (int d = lane; d < 256; d += 32) {
            unsigned int ga = grow[d], ra = rrow[d];
            s += bf2f((unsigned short)(ga & 0xFFFFu)) * bf2f((unsigned short)(ra & 0xFFFFu));
            s += bf2f((unsigned short)(ga >> 16)) * bf2f((unsigned short)(ra >> 16));
        }
        for (int m = 16; m >= 1; m >>= 1) s += __shfl_xor(s, m, 32);
        if (lane == 0) sS[n] = s + isc[brow + sL[n]] + isc[brow + sR[n]];
    }
    __syncthreads();

    if (tid == 0) {
        float mx = -3.4e38f;
        for (int n = 0; n < N; ++n) mx = fmaxf(mx, sS[n]);
        float sum = 0.f;
        for (int n = 0; n < N; ++n) { float e = expf(sS[n] - mx); sPr[n] = e; sum += e; }
        float inv = 1.f / sum, outs = 0.f;
        for (int n = 0; n < N; ++n) { sPr[n] *= inv; outs += sPr[n] * sS[n]; }
        sMisc[0] = outs;
    }
    __syncthreads();

    level_gemm_reduce<MT>(A32, (const uint4*)Wp, b_in, sPr, sOut, lane, wave);
    __syncthreads();

    // unit norm
    {
        float ss = 0.f;
        for (int idx = tid; idx < 512; idx += 256) { float v = sOut[idx]; ss += v * v; }
        for (int m = 16; m >= 1; m >>= 1) ss += __shfl_xor(ss, m, 32);
        if (lane == 0) red[wave] = ss;
        __syncthreads();
        if (tid == 0) {
            float t = 0.f;
            for (int w = 0; w < 8; ++w) t += red[w];
            sMisc[1] = 1.f / fmaxf(sqrtf(t), EPSF);
        }
    }
    __syncthreads();

    int ocell = off_of(level) + i;
    size_t orow = brow + ocell;
    float inv = sMisc[1];
    for (int idx = tid; idx < 512; idx += 256) {
        float v = sOut[idx] * inv;
        outbuf[orow * 1024 + idx] = v;
        ih_bf[orow * 512 + idx] = f2bf(v);
    }
    if (tid == 0) isc[orow] = sMisc[0];
    __syncthreads();
    // fused GEMV: g = h @ S_in
    for (int j = tid; j < 512; j += 256) {
        const unsigned short* st = SinT + (size_t)j * 512;
        float s = 0.f;
        for (int k = 0; k < 512; ++k) s += sOut[k] * bf2f(st[k]);
        s *= inv;
        g_in[orow * 512 + j] = f2bf(s);
    }
}

// ---------------------------------------------------------------------------
// Outside level: one WG per (b, i). Candidates n=0..N-1 (N=TT-level-1).
// ---------------------------------------------------------------------------
template <int MT>
__global__ void __launch_bounds__(256) outside_level_kernel(
    int level, const unsigned short* __restrict__ ih_bf,
    unsigned short* __restrict__ oh_bf, const unsigned short* __restrict__ g_out,
    const float* __restrict__ isc, float* __restrict__ osc,
    const unsigned short* __restrict__ Wp, const float* __restrict__ b_out,
    float* __restrict__ outbuf) {
    const int L = TT - level, N = TT - level - 1;
    constexpr int Npad = MT * 16;
    int b = blockIdx.x / L, i = blockIdx.x % L;

    extern __shared__ char smem[];
    unsigned int* A32 = (unsigned int*)smem;
    float* sOut = (float*)(smem + (size_t)Npad * 2048);
    float* sS = sOut + 512;
    float* sPr = sS + 48;
    int* sQ = (int*)(sPr + 48);   // sibling (inside) cell
    int* sPc = sQ + 48;           // parent (outside) cell
    float* sMisc = (float*)(sPc + 48);
    __shared__ float red[8];

    int tid = threadIdx.x, lane = tid & 31, wave = tid >> 5;
    size_t brow = (size_t)b * NCELLS;

    if (tid < Npad) {
        if (tid < N) {
            int n = tid, p, q;
            if (n < i) { int s = n + 1; p = off_of(level + s) + i - s; q = off_of(s - 1) + i - s; }
            else { int s = n - i + 1; p = off_of(level + s) + i; q = off_of(s - 1) + i + level + 1; }
            sQ[tid] = q; sPc[tid] = p;
        }
        sPr[tid] = 0.f;
    }
    for (int idx = tid; idx < 512; idx += 256) sOut[idx] = 0.f;
    __syncthreads();

    const unsigned int* ih32 = (const unsigned int*)ih_bf;
    const unsigned int* oh32 = (const unsigned int*)oh_bf;
    for (int idx = tid; idx < Npad * 512; idx += 256) {
        int n = idx >> 9, o = idx & 511;
        unsigned int v = 0u;
        if (n < N) {
            v = (o < 256) ? ih32[(brow + sQ[n]) * 256 + o]
                          : oh32[(brow + sPc[n]) * 256 + (o - 256)];
        }
        A32[idx] = v;
    }
    __syncthreads();

    const unsigned int* g32 = (const unsigned int*)g_out;
    for (int n = wave; n < N; n += 8) {
        const unsigned int* grow = g32 + (brow + sQ[n]) * 256;
        const unsigned int* rrow = A32 + n * 512 + 256;   // ph half
        float s = 0.f;
        for (int d = lane; d < 256; d += 32) {
            unsigned int ga = grow[d], ra = rrow[d];
            s += bf2f((unsigned short)(ga & 0xFFFFu)) * bf2f((unsigned short)(ra & 0xFFFFu));
            s += bf2f((unsigned short)(ga >> 16)) * bf2f((unsigned short)(ra >> 16));
        }
        for (int m = 16; m >= 1; m >>= 1) s += __shfl_xor(s, m, 32);
        if (lane == 0) sS[n] = s + isc[brow + sQ[n]] + osc[brow + sPc[n]];
    }
    __syncthreads();

    if (tid == 0) {
        float mx = -3.4e38f;
        for (int n = 0; n < N; ++n) mx = fmaxf(mx, sS[n]);
        float sum = 0.f;
        for (int n = 0; n < N; ++n) { float e = expf(sS[n] - mx); sPr[n] = e; sum += e; }
        float inv = 1.f / sum, outs = 0.f;
        for (int n = 0; n < N; ++n) { sPr[n] *= inv; outs += sPr[n] * sS[n]; }
        sMisc[0] = outs;
    }
    __syncthreads();

    level_gemm_reduce<MT>(A32, (const uint4*)Wp, b_out, sPr, sOut, lane, wave);
    __syncthreads();

    {
        float ss = 0.f;
        for (int idx = tid; idx < 512; idx += 256) { float v = sOut[idx]; ss += v * v; }
        for (int m = 16; m >= 1; m >>= 1) ss += __shfl_xor(ss, m, 32);
        if (lane == 0) red[wave] = ss;
        __syncthreads();
        if (tid == 0) {
            float t = 0.f;
            for (int w = 0; w < 8; ++w) t += red[w];
            sMisc[1] = 1.f / fmaxf(sqrtf(t), EPSF);
        }
    }
    __syncthreads();

    int ocell = off_of(level) + i;
    size_t orow = brow + ocell;
    float inv = sMisc[1];
    for (int idx = tid; idx < 512; idx += 256) {
        float v = sOut[idx] * inv;
        outbuf[orow * 1024 + 512 + idx] = v;
        oh_bf[orow * 512 + idx] = f2bf(v);
    }
    if (tid == 0) osc[orow] = sMisc[0];
}

// ---------------------------------------------------------------------------
extern "C" void kernel_launch(void* const* d_in, const int* in_sizes, int n_in,
                              void* d_out, int out_size, void* d_ws, size_t ws_size,
                              hipStream_t stream) {
    (void)in_sizes; (void)n_in; (void)out_size; (void)ws_size;
    const float* x      = (const float*)d_in[0];
    const float* W_leaf = (const float*)d_in[1];
    const float* b_leaf = (const float*)d_in[2];
    const float* W_in   = (const float*)d_in[3];
    const float* b_in   = (const float*)d_in[4];
    const float* S_in   = (const float*)d_in[5];
    const float* W_out  = (const float*)d_in[6];
    const float* b_out  = (const float*)d_in[7];
    const float* S_out  = (const float*)d_in[8];
    const float* root_h = (const float*)d_in[9];
    float* out = (float*)d_out;

    char* ws = (char*)d_ws;
    size_t off = 0;
    auto take = [&](size_t bytes) { char* p = ws + off; off = (off + bytes + 255) & ~(size_t)255; return p; };
    const size_t cellmat = (size_t)BB * NCELLS * DD * 2;   // bf16
    unsigned short* ih_bf  = (unsigned short*)take(cellmat);
    unsigned short* oh_bf  = (unsigned short*)take(cellmat);
    unsigned short* g_in   = (unsigned short*)take(cellmat);
    unsigned short* g_out  = (unsigned short*)take(cellmat);
    float* isc             = (float*)take((size_t)BB * NCELLS * 4);
    float* osc             = (float*)take((size_t)BB * NCELLS * 4);
    unsigned short* Wleaf_p = (unsigned short*)take((size_t)DD * DD * 2);
    unsigned short* Win_p   = (unsigned short*)take((size_t)2 * DD * DD * 2);
    unsigned short* Wout_p  = (unsigned short*)take((size_t)2 * DD * DD * 2);
    unsigned short* Sout_p  = (unsigned short*)take((size_t)DD * DD * 2);
    unsigned short* SinT    = (unsigned short*)take((size_t)DD * DD * 2);

    // 1) weight prep
    {
        int th512 = (DD / 32) * (DD / 16) * 32;         // 16384
        int th1024 = (2 * DD / 32) * (DD / 16) * 32;    // 32768
        pack_weight_kernel<<<(th512 + 255) / 256, 256, 0, stream>>>(W_leaf, Wleaf_p, DD, DD);
        pack_weight_kernel<<<(th1024 + 255) / 256, 256, 0, stream>>>(W_in, Win_p, 2 * DD, DD);
        pack_weight_kernel<<<(th1024 + 255) / 256, 256, 0, stream>>>(W_out, Wout_p, 2 * DD, DD);
        pack_weight_kernel<<<(th512 + 255) / 256, 256, 0, stream>>>(S_out, Sout_p, DD, DD);
        transpose_bf16_kernel<<<(DD * DD + 255) / 256, 256, 0, stream>>>(S_in, SinT);
    }
    // 2) init scores + root
    zero2_kernel<<<(BB * NCELLS + 255) / 256, 256, 0, stream>>>(isc, osc, BB * NCELLS);
    root_kernel<<<1, 512, 0, stream>>>(root_h, out, oh_bf);

    // 3) leaf
    {
        size_t smem = 16 * 512 * 2 + 16 * 512 * 4 + 64;
        leaf_kernel<<<(BB * TT) / 16, 256, smem, stream>>>(x, Wleaf_p, b_leaf, SinT, out, ih_bf, g_in);
    }
    // 4) inside pass
    for (int level = 1; level < TT; ++level) {
        int L = TT - level, N = level;
        int mt = (N + 15) / 16;
        size_t smem = (size_t)(mt * 16) * 2048 + 2880;
        switch (mt) {
        case 1: inside_level_kernel<1><<<BB * L, 256, smem, stream>>>(level, ih_bf, g_in, isc, Win_p, b_in, SinT, out); break;
        case 2: inside_level_kernel<2><<<BB * L, 256, smem, stream>>>(level, ih_bf, g_in, isc, Win_p, b_in, SinT, out); break;
        default: inside_level_kernel<3><<<BB * L, 256, smem, stream>>>(level, ih_bf, g_in, isc, Win_p, b_in, SinT, out); break;
        }
    }
    // 5) g_out = ih @ S_out for all cells
    gout_kernel<<<(BB * NCELLS) / 16, 256, 16 * 512 * 2, stream>>>(ih_bf, Sout_p, g_out);
    // 6) outside pass
    for (int level = TT - 2; level >= 0; --level) {
        int L = TT - level, N = TT - level - 1;
        int mt = (N + 15) / 16;
        size_t smem = (size_t)(mt * 16) * 2048 + 2880;
        switch (mt) {
        case 1: outside_level_kernel<1><<<BB * L, 256, smem, stream>>>(level, ih_bf, oh_bf, g_out, isc, osc, Wout_p, b_out, out); break;
        case 2: outside_level_kernel<2><<<BB * L, 256, smem, stream>>>(level, ih_bf, oh_bf, g_out, isc, osc, Wout_p, b_out, out); break;
        default: outside_level_kernel<3><<<BB * L, 256, smem, stream>>>(level, ih_bf, oh_bf, g_out, isc, osc, Wout_p, b_out, out); break;
        }
    }
}
